// LRSA_2018634629368
// MI455X (gfx1250) — matrix operand order repaired and checked by reading, not compiled, not run on hardware
//
#include <hip/hip_runtime.h>
#include <hip/hip_bf16.h>
#include <cstdint>

// ---------------------------------------------------------------------------
// Problem constants (from setup_inputs): b=2, dim=96, h=w=256, ps=16
// step=14, 19x19=361 windows/image, 722 windows total, 256 tokens/window
// qk_dim=64, heads=4, hd_q=16, hd_v=24, mlp=192
// ---------------------------------------------------------------------------
#define BATCH   2
#define CDIM    96
#define HDIM    256
#define WDIM    256
#define HW      65536
#define NWIN1D  19
#define NWIN    361
#define BN      722          // BATCH * NWIN
#define SEQ     256          // ps*ps tokens per window
#define QK      64
#define HEADS   4
#define HDQ     16
#define HDV     24
#define MLP     192
#define QKV     224          // QK + QK + CDIM (fused projection width)
#define M1      (BN * SEQ)   // 184832 rows for attention-side GEMMs
#define M2      (BATCH * HW) // 131072 rows for MLP-side GEMMs

typedef _Float16 half16 __attribute__((ext_vector_type(16)));
typedef float    float8 __attribute__((ext_vector_type(8)));

struct __align__(16) U4 { unsigned a, b, c, d; };
union AFrag { half16 v; _Float16 h[16]; U4 q[2]; };

__device__ __forceinline__ float gelu_exact(float v) {
    return 0.5f * v * (1.f + erff(v * 0.70710678118654752f));
}

// ---------------------------------------------------------------------------
// f32[K,N] -> f16[N,K] transposed weight conversion (weights are tiny)
// ---------------------------------------------------------------------------
__global__ __launch_bounds__(256)
void cvtT_kernel(const float* __restrict__ s, _Float16* __restrict__ d,
                 int K, int N) {
    int i = blockIdx.x * 256 + threadIdx.x;
    if (i < K * N) {
        int k = i / N, n = i % N;
        d[n * K + k] = (_Float16)s[i];
    }
}

// ---------------------------------------------------------------------------
// Window extraction + LayerNorm1. One warp (32 lanes) per token; lane c owns
// channels {c, c+32, c+64}.
// ---------------------------------------------------------------------------
__global__ __launch_bounds__(256)
void winln1_kernel(const float* __restrict__ x, const float* __restrict__ w,
                   const float* __restrict__ bb, float* __restrict__ tok,
                   _Float16* __restrict__ yln) {
    int gt = blockIdx.x * 8 + (threadIdx.x >> 5);
    int lane = threadIdx.x & 31;
    if (gt >= M1) return;
    int wdw = gt >> 8, t = gt & 255;
    int b = wdw / NWIN, wi = wdw % NWIN;
    int wy = wi / NWIN1D, wx = wi % NWIN1D;
    int top = min(wy * 14, 240), left = min(wx * 14, 240);
    int y = top + (t >> 4), xq = left + (t & 15);
    const float* px = x + (size_t)b * CDIM * HW + (size_t)y * WDIM + xq;
    float v0 = px[(size_t)lane * HW];
    float v1 = px[(size_t)(lane + 32) * HW];
    float v2 = px[(size_t)(lane + 64) * HW];
    float s = v0 + v1 + v2;
    float s2 = v0 * v0 + v1 * v1 + v2 * v2;
#pragma unroll
    for (int off = 16; off >= 1; off >>= 1) {
        s  += __shfl_xor(s, off, 32);
        s2 += __shfl_xor(s2, off, 32);
    }
    float mean = s * (1.f / 96.f);
    float var = s2 * (1.f / 96.f) - mean * mean;
    float rstd = rsqrtf(var + 1e-5f);
    size_t base = (size_t)gt * CDIM;
    tok[base + lane]      = v0;
    tok[base + lane + 32] = v1;
    tok[base + lane + 64] = v2;
    yln[base + lane]      = (_Float16)((v0 - mean) * rstd * w[lane]      + bb[lane]);
    yln[base + lane + 32] = (_Float16)((v1 - mean) * rstd * w[lane + 32] + bb[lane + 32]);
    yln[base + lane + 64] = (_Float16)((v2 - mean) * rstd * w[lane + 64] + bb[lane + 64]);
}

// ---------------------------------------------------------------------------
// WMMA GEMM: C[M,N] = A[M,K](f16) * B[K,N] (+bias[N]) (+ACT=gelu) (+resid)
// B supplied PRE-TRANSPOSED as Bt[N,K] f16 so both LDS tiles stage with b128.
// Block tile 128x64, 8 waves (4 row groups x 2 col groups), each wave 32x32.
// MODE 0: store to outF/outH [M,N]
// MODE 1: window overlap-add: atomicAdd into outF = acc (b,c,256,256)
// MODE 2: transposed store: outF = out (b,c,h,w), row encodes (b,hw)
// MODE 3: fused QKV split store: outH=q [M,64], outH2=k [M,64], outH3=v [M,96]
// ---------------------------------------------------------------------------
template <int N, int K, int MODE, int ACT>
__global__ __launch_bounds__(256)
void gemm_kernel(const _Float16* __restrict__ A, const _Float16* __restrict__ Bt,
                 float* __restrict__ outF, _Float16* __restrict__ outH,
                 _Float16* __restrict__ outH2, _Float16* __restrict__ outH3,
                 const float* __restrict__ bias, const float* __restrict__ resid) {
    __shared__ _Float16 As[128 * 32];   // [m][k]
    __shared__ _Float16 Bs[64 * 32];    // [n][k]
    const int tid = threadIdx.x;
    const int lane = tid & 31, wave = tid >> 5;
    const int wr = wave & 3, wc = wave >> 2;
    const int m0 = blockIdx.y * 128;
    const int n0 = blockIdx.x * 64;

    float8 acc[2][2] = {};
#pragma unroll
    for (int k0 = 0; k0 < K; k0 += 32) {
        // A tile: 128 rows x 32 halves = 512 b128 chunks, 2 per thread
#pragma unroll
        for (int it = 0; it < 2; ++it) {
            int idx = tid + it * 256;
            int row = idx >> 2, u = idx & 3;
            *(U4*)(As + row * 32 + u * 8) =
                *(const U4*)(A + (size_t)(m0 + row) * K + k0 + u * 8);
        }
        // B tile: 64 rows x 32 halves = 256 b128 chunks, 1 per thread
        {
            int n = tid >> 2, u = tid & 3;
            U4 val = {0u, 0u, 0u, 0u};
            if (n0 + n < N)
                val = *(const U4*)(Bt + (size_t)(n0 + n) * K + k0 + u * 8);
            *(U4*)(Bs + n * 32 + u * 8) = val;
        }
        // Prefetch next A k-tile while this one is consumed.
        if (k0 + 32 < K)
            __builtin_prefetch(A + (size_t)(m0 + (tid >> 1)) * K + k0 + 32, 0, 0);
        __syncthreads();

        AFrag a[2], b[2];
        const int koff = (lane >> 4) * 8;
        const int arow = wr * 32 + (lane & 15);
#pragma unroll
        for (int rr = 0; rr < 2; ++rr) {
            const _Float16* ap = As + (size_t)(arow + rr * 16) * 32;
            a[rr].q[0] = *(const U4*)(ap + koff);       // k = koff..koff+7
            a[rr].q[1] = *(const U4*)(ap + 16 + koff);  // k = 16+koff..
        }
        const int bk = (lane >> 4) * 8;
#pragma unroll
        for (int cc = 0; cc < 2; ++cc) {
            const _Float16* bp =
                Bs + (size_t)(wc * 32 + cc * 16 + (lane & 15)) * 32 + bk;
            b[cc].q[0] = *(const U4*)(bp);
            b[cc].q[1] = *(const U4*)(bp + 16);
        }
#pragma unroll
        for (int rr = 0; rr < 2; ++rr)
#pragma unroll
            for (int cc = 0; cc < 2; ++cc)
                acc[rr][cc] = __builtin_amdgcn_wmma_f32_16x16x32_f16(
                    false, a[rr].v, false, b[cc].v, (short)0, acc[rr][cc],
                    false, false);
        __syncthreads();
    }

    const int row0 = m0 + wr * 32;
    const int col0 = n0 + wc * 32;
#pragma unroll
    for (int rr = 0; rr < 2; ++rr)
#pragma unroll
        for (int cc = 0; cc < 2; ++cc) {
            int col = col0 + cc * 16 + (lane & 15);
            if (col >= N) continue;
#pragma unroll
            for (int r = 0; r < 8; ++r) {
                int row = row0 + rr * 16 + (lane >> 4) * 8 + r;
                float vv = acc[rr][cc][r];
                if (bias)  vv += bias[col];
                if (ACT == 1) vv = gelu_exact(vv);
                if (resid) vv += resid[(size_t)row * N + col];
                if (MODE == 0) {
                    size_t off = (size_t)row * N + col;
                    if (outF) outF[off] = vv;
                    if (outH) outH[off] = (_Float16)vv;
                } else if (MODE == 1) {
                    // row = window*256 + token; scatter into (b,c,y,x) with
                    // atomic overlap-add (patch_reverse accumulate).
                    int wdw = row >> 8, t = row & 255;
                    int b2 = wdw / NWIN, wi = wdw % NWIN;
                    int wy = wi / NWIN1D, wx = wi % NWIN1D;
                    int top = min(wy * 14, 240), left = min(wx * 14, 240);
                    int y = top + (t >> 4), xq = left + (t & 15);
                    atomicAdd(outF + (((size_t)b2 * CDIM + col) * HDIM + y) * WDIM + xq,
                              vv);
                } else if (MODE == 2) {  // row = b*HW + hw -> out[b][col][hw]
                    outF[((size_t)(row >> 16) * CDIM + col) * HW + (row & 65535)] = vv;
                } else {                 // MODE 3: split q | k | v stores
                    _Float16 hv = (_Float16)vv;
                    if (col < QK)
                        outH[(size_t)row * QK + col] = hv;
                    else if (col < 2 * QK)
                        outH2[(size_t)row * QK + (col - QK)] = hv;
                    else
                        outH3[(size_t)row * CDIM + (col - 2 * QK)] = hv;
                }
            }
        }
}

// ---------------------------------------------------------------------------
// Fused attention: one block per (window, head). 8 waves x 32 query rows.
// Flash-style online softmax over 8 tiles of 32 keys, WMMA for QK^T and P*V.
// - V staged transposed (Vt[vcol][key]) so P*V B-fragments load as b128.
// - Vt row 24 is all-ones: the P*V WMMA accumulates the softmax denominator
//   in output column 24 (rescaled by the same alpha as the numerator).
// - Online max kept per 8-row group (one shuffle-reduce per rr per tile).
// Q/K f16 [BN,256,64], V f16 [BN,256,96]; output f16 [BN,256,96].
// ---------------------------------------------------------------------------
__global__ __launch_bounds__(256)
void attn_kernel(const _Float16* __restrict__ Q, const _Float16* __restrict__ Kx,
                 const _Float16* __restrict__ V, _Float16* __restrict__ O) {
    __shared__ _Float16 Ks[SEQ * 16];      // [key][d]
    __shared__ _Float16 Vt[32 * SEQ];      // [vcol][key]; row24=1, 25..31=0
    __shared__ _Float16 Ps[8][32 * 32];    // per-wave P scratch [row][key]
    const int blk = blockIdx.x;
    const int win = blk >> 2, head = blk & 3;
    const int tid = threadIdx.x, lane = tid & 31, wave = tid >> 5;

    {   // stage K (b128) and V (transposed) for this head
        int key = tid;
        const _Float16* ksrc = Kx + ((size_t)win * SEQ + key) * QK + head * HDQ;
        *(U4*)(Ks + key * 16)     = *(const U4*)(ksrc);
        *(U4*)(Ks + key * 16 + 8) = *(const U4*)(ksrc + 8);
        const _Float16* vsrc = V + ((size_t)win * SEQ + key) * CDIM + head * HDV;
        union { U4 q[3]; _Float16 h[24]; } vt;
        vt.q[0] = *(const U4*)(vsrc);
        vt.q[1] = *(const U4*)(vsrc + 8);
        vt.q[2] = *(const U4*)(vsrc + 16);
#pragma unroll
        for (int n = 0; n < HDV; ++n) Vt[n * SEQ + key] = vt.h[n];
        Vt[HDV * SEQ + key] = (_Float16)1.f;      // ones row -> denominator
#pragma unroll
        for (int n = HDV + 1; n < 32; ++n) Vt[n * SEQ + key] = (_Float16)0.f;
    }
    __syncthreads();

    const int koff = (lane >> 4) * 8;
    const int qrow0 = wave * 32;
    AFrag qa[2];
#pragma unroll
    for (int rr = 0; rr < 2; ++rr) {
        const _Float16* qsrc =
            Q + ((size_t)win * SEQ + qrow0 + rr * 16 + (lane & 15)) * QK + head * HDQ;
        qa[rr].q[0] = *(const U4*)(qsrc + koff);     // dims koff..koff+7 (<16)
#pragma unroll
        for (int j = 0; j < 8; ++j) qa[rr].h[j] *= (_Float16)0.25f;  // 1/sqrt(16)
#pragma unroll
        for (int j = 8; j < 16; ++j) qa[rr].h[j] = (_Float16)0.f;    // K-pad
    }

    // kb pad lanes (gemm-K 16..31) stay zero across all tiles.
    AFrag kb[2];
#pragma unroll
    for (int kk = 0; kk < 2; ++kk)
#pragma unroll
        for (int j = 0; j < 16; ++j) kb[kk].h[j] = (_Float16)0.f;

    float mg[2] = {-1e30f, -1e30f};   // per 8-row-group running max
    float8 o[2][2] = {};

    for (int t = 0; t < 8; ++t) {           // 32 keys per tile
        {                                   // gemm-K (dims) koff..koff+7
#pragma unroll
            for (int kk = 0; kk < 2; ++kk) {
                const _Float16* ks = Ks + (size_t)(t * 32 + kk * 16 + (lane & 15)) * 16;
                kb[kk].q[0] = *(const U4*)(ks + koff);
            }
        }
        float8 zero8 = {};
        float8 s[2][2];
#pragma unroll
        for (int rr = 0; rr < 2; ++rr)
#pragma unroll
            for (int kk = 0; kk < 2; ++kk)
                s[rr][kk] = __builtin_amdgcn_wmma_f32_16x16x32_f16(
                    false, qa[rr].v, false, kb[kk].v, (short)0, zero8, false, false);

#pragma unroll
        for (int rr = 0; rr < 2; ++rr) {
            // group max over this lane-half's 8 rows x 32 keys
            float mx = -1e30f;
#pragma unroll
            for (int r = 0; r < 8; ++r)
                mx = fmaxf(mx, fmaxf(s[rr][0][r], s[rr][1][r]));
#pragma unroll
            for (int off = 8; off >= 1; off >>= 1)
                mx = fmaxf(mx, __shfl_xor(mx, off, 32));
            float nm = fmaxf(mg[rr], mx);
            float al = __expf(mg[rr] - nm);
            mg[rr] = nm;
#pragma unroll
            for (int r = 0; r < 8; ++r) {
                float p0 = __expf(s[rr][0][r] - nm);
                float p1 = __expf(s[rr][1][r] - nm);
                o[rr][0][r] *= al;
                o[rr][1][r] *= al;
                int row = rr * 16 + (lane >> 4) * 8 + r;
                Ps[wave][row * 32 + (lane & 15)]      = (_Float16)p0;
                Ps[wave][row * 32 + 16 + (lane & 15)] = (_Float16)p1;
            }
        }
        asm volatile("s_wait_dscnt 0" ::: "memory");   // same-wave LDS RAW

        AFrag pa[2], vb[2];
#pragma unroll
        for (int rr = 0; rr < 2; ++rr) {
            const _Float16* pp = Ps[wave] + (size_t)(rr * 16 + (lane & 15)) * 32;
            pa[rr].q[0] = *(const U4*)(pp + koff);        // keys koff..+7
            pa[rr].q[1] = *(const U4*)(pp + 16 + koff);   // keys 16+koff..
        }
        const int bk = (lane >> 4) * 8;
#pragma unroll
        for (int cc = 0; cc < 2; ++cc) {
            const _Float16* vp =
                Vt + (size_t)(cc * 16 + (lane & 15)) * SEQ + t * 32 + bk;
            vb[cc].q[0] = *(const U4*)(vp);
            vb[cc].q[1] = *(const U4*)(vp + 16);
        }
#pragma unroll
        for (int rr = 0; rr < 2; ++rr)
#pragma unroll
            for (int cc = 0; cc < 2; ++cc)
                o[rr][cc] = __builtin_amdgcn_wmma_f32_16x16x32_f16(
                    false, pa[rr].v, false, vb[cc].v, (short)0, o[rr][cc],
                    false, false);
    }

    // Denominator lives in output column 24 (frag cc=1, lane&15 == 8).
#pragma unroll
    for (int rr = 0; rr < 2; ++rr) {
        float lrec[8];
#pragma unroll
        for (int r = 0; r < 8; ++r)
            lrec[r] = 1.f / __shfl(o[rr][1][r], (lane & 16) + 8, 32);
#pragma unroll
        for (int cc = 0; cc < 2; ++cc) {
            int vcol = cc * 16 + (lane & 15);
            if (vcol >= HDV) continue;
#pragma unroll
            for (int r = 0; r < 8; ++r) {
                int row = qrow0 + rr * 16 + (lane >> 4) * 8 + r;
                O[((size_t)win * SEQ + row) * CDIM + head * HDV + vcol] =
                    (_Float16)(o[rr][cc][r] * lrec[r]);
            }
        }
    }
}

__device__ __forceinline__ float divfac(int y) {
    float f = 1.f;
    for (int i = 14; i < 254; i += 14) {
        int top = i;
        if (top + 16 > 256) top = 240;
        int down = i + 2;
        if (y >= top && y < down) f *= 0.5f;
    }
    return f;
}

// ---------------------------------------------------------------------------
// div-factor scaling + LayerNorm2. Warp per pixel.
// ---------------------------------------------------------------------------
__global__ __launch_bounds__(256)
void lnscale_kernel(const float* __restrict__ acc, const float* __restrict__ w,
                    const float* __restrict__ bb, float* __restrict__ tok2,
                    _Float16* __restrict__ y2) {
    int p = blockIdx.x * 8 + (threadIdx.x >> 5);
    int lane = threadIdx.x & 31;
    if (p >= M2) return;
    int b = p >> 16, hw = p & 65535;
    int y = hw >> 8, xq = hw & 255;
    float f = divfac(y) * divfac(xq);
    const float* pa = acc + (size_t)b * CDIM * HW + hw;
    float v0 = pa[(size_t)lane * HW] * f;
    float v1 = pa[(size_t)(lane + 32) * HW] * f;
    float v2 = pa[(size_t)(lane + 64) * HW] * f;
    float s = v0 + v1 + v2;
    float s2 = v0 * v0 + v1 * v1 + v2 * v2;
#pragma unroll
    for (int off = 16; off >= 1; off >>= 1) {
        s  += __shfl_xor(s, off, 32);
        s2 += __shfl_xor(s2, off, 32);
    }
    float mean = s * (1.f / 96.f);
    float var = s2 * (1.f / 96.f) - mean * mean;
    float rstd = rsqrtf(var + 1e-5f);
    size_t base = (size_t)p * CDIM;
    tok2[base + lane]      = v0;
    tok2[base + lane + 32] = v1;
    tok2[base + lane + 64] = v2;
    y2[base + lane]      = (_Float16)((v0 - mean) * rstd * w[lane]      + bb[lane]);
    y2[base + lane + 32] = (_Float16)((v1 - mean) * rstd * w[lane + 32] + bb[lane + 32]);
    y2[base + lane + 64] = (_Float16)((v2 - mean) * rstd * w[lane + 64] + bb[lane + 64]);
}

// ---------------------------------------------------------------------------
// Depthwise 5x5 conv + GELU + residual -> f16. h1 layout (b, hw, 192) holds
// gelu(fc1) already. Consecutive tid -> consecutive channel: coalesced taps.
// ---------------------------------------------------------------------------
__global__ __launch_bounds__(256)
void dwconv_kernel(const float* __restrict__ h1, const float* __restrict__ wgt,
                   const float* __restrict__ bias, _Float16* __restrict__ h2) {
    size_t i = (size_t)blockIdx.x * 256 + threadIdx.x;
    if (i >= (size_t)M2 * MLP) return;
    int ch = (int)(i % MLP);
    size_t j = i / MLP;
    int hw = (int)(j % HW);
    int b = (int)(j / HW);
    int y = hw >> 8, xq = hw & 255;
    float s = 0.f;
    const float* wz = wgt + ch * 25;
    for (int dy = -2; dy <= 2; ++dy) {
        int yy = y + dy;
        if (yy < 0 || yy > 255) continue;
        for (int dx = -2; dx <= 2; ++dx) {
            int xx = xq + dx;
            if (xx < 0 || xx > 255) continue;
            s += h1[((size_t)b * HW + (size_t)yy * 256 + xx) * MLP + ch] *
                 wz[(dy + 2) * 5 + (dx + 2)];
        }
    }
    s += bias[ch];
    h2[i] = (_Float16)(h1[i] + gelu_exact(s));
}

// ---------------------------------------------------------------------------
extern "C" void kernel_launch(void* const* d_in, const int* in_sizes, int n_in,
                              void* d_out, int out_size, void* d_ws, size_t ws_size,
                              hipStream_t stream) {
    const float* x       = (const float*)d_in[0];
    const float* norm1_w = (const float*)d_in[1];
    const float* norm1_b = (const float*)d_in[2];
    const float* wq      = (const float*)d_in[3];
    const float* wk      = (const float*)d_in[4];
    const float* wv      = (const float*)d_in[5];
    const float* wproj   = (const float*)d_in[6];
    const float* norm2_w = (const float*)d_in[7];
    const float* norm2_b = (const float*)d_in[8];
    const float* fc1_w   = (const float*)d_in[9];
    const float* fc1_b   = (const float*)d_in[10];
    const float* dw_w    = (const float*)d_in[11];
    const float* dw_b    = (const float*)d_in[12];
    const float* fc2_w   = (const float*)d_in[13];
    const float* fc2_b   = (const float*)d_in[14];
    (void)in_sizes; (void)n_in; (void)out_size; (void)ws_size;

    char* ws = (char*)d_ws;
    auto al = [](size_t v) { return (v + 255) & ~(size_t)255; };

    // ---- Phase 1 arena (attention side) ----
    size_t o = 0;
    auto take1 = [&](size_t bytes) { size_t r = o; o = al(o + bytes); return r; };
    size_t o_tok  = take1((size_t)M1 * CDIM * 4);   // f32 residual tokens
    size_t o_yln  = take1((size_t)M1 * CDIM * 2);   // f16 LN1 output
    size_t o_q    = take1((size_t)M1 * QK * 2);
    size_t o_k    = take1((size_t)M1 * QK * 2);
    size_t o_v    = take1((size_t)M1 * CDIM * 2);
    size_t o_attn = take1((size_t)M1 * CDIM * 2);
    size_t end1 = o;

    // acc lives above phase 1 (concurrently live with tok/attn during proj)
    size_t o_acc = end1;
    size_t acc_end = al(o_acc + (size_t)BATCH * CDIM * HW * 4);

    // ---- Phase 2 arena (MLP side), aliases dead phase-1 buffers ----
    size_t o2 = 0;
    auto take2 = [&](size_t bytes) { size_t r = o2; o2 = al(o2 + bytes); return r; };
    size_t o_tok2 = take2((size_t)M2 * CDIM * 4);
    size_t o_y2   = take2((size_t)M2 * CDIM * 2);
    size_t o_h1   = take2((size_t)M2 * MLP * 4);
    size_t o_h2   = take2((size_t)M2 * MLP * 2);
    size_t end2 = o2;

    // ---- f16 transposed weights [N,K], above everything ----
    size_t ow = al(acc_end > end2 ? acc_end : end2);
    auto takew = [&](size_t bytes) { size_t r = ow; ow = al(ow + bytes); return r; };
    size_t o_wqkv = takew((size_t)QKV * CDIM * 2);   // wq^T | wk^T | wv^T
    size_t o_wp16 = takew((size_t)CDIM * CDIM * 2);
    size_t o_f116 = takew((size_t)CDIM * MLP * 2);
    size_t o_f216 = takew((size_t)MLP * CDIM * 2);

    _Float16* wqkv = (_Float16*)(ws + o_wqkv);
    _Float16* wp16 = (_Float16*)(ws + o_wp16);
    _Float16* f116 = (_Float16*)(ws + o_f116);
    _Float16* f216 = (_Float16*)(ws + o_f216);

    // concatenated transposed projection weights: rows 0..63 q, 64..127 k,
    // 128..223 v
    cvtT_kernel<<<(CDIM * QK + 255) / 256, 256, 0, stream>>>(
        wq, wqkv, CDIM, QK);
    cvtT_kernel<<<(CDIM * QK + 255) / 256, 256, 0, stream>>>(
        wk, wqkv + (size_t)QK * CDIM, CDIM, QK);
    cvtT_kernel<<<(CDIM * CDIM + 255) / 256, 256, 0, stream>>>(
        wv, wqkv + (size_t)2 * QK * CDIM, CDIM, CDIM);
    cvtT_kernel<<<(CDIM * CDIM + 255) / 256, 256, 0, stream>>>(wproj, wp16, CDIM, CDIM);
    cvtT_kernel<<<(CDIM * MLP + 255) / 256, 256, 0, stream>>>(fc1_w, f116, CDIM, MLP);
    cvtT_kernel<<<(MLP * CDIM + 255) / 256, 256, 0, stream>>>(fc2_w, f216, MLP, CDIM);

    float*    tok  = (float*)(ws + o_tok);
    _Float16* yln  = (_Float16*)(ws + o_yln);
    _Float16* qb   = (_Float16*)(ws + o_q);
    _Float16* kb   = (_Float16*)(ws + o_k);
    _Float16* vb   = (_Float16*)(ws + o_v);
    _Float16* attn = (_Float16*)(ws + o_attn);
    float*    acc  = (float*)(ws + o_acc);

    // 1. windows + LN1
    winln1_kernel<<<M1 / 8, 256, 0, stream>>>(x, norm1_w, norm1_b, tok, yln);

    // 2. fused Q|K|V projection: one pass over yln
    gemm_kernel<QKV, CDIM, 3, 0><<<dim3(4, M1 / 128), 256, 0, stream>>>(
        yln, wqkv, nullptr, qb, kb, vb, nullptr, nullptr);

    // zero overlap-add accumulator (must complete before proj GEMM)
    hipMemsetAsync(acc, 0, (size_t)BATCH * CDIM * HW * 4, stream);

    // 3. fused flash attention per (window, head)
    attn_kernel<<<BN * HEADS, 256, 0, stream>>>(qb, kb, vb, attn);

    // 4. output projection + residual tok, fused overlap-add scatter into acc
    gemm_kernel<CDIM, CDIM, 1, 0><<<dim3(2, M1 / 128), 256, 0, stream>>>(
        attn, wp16, acc, nullptr, nullptr, nullptr, nullptr, tok);

    // 5. div factors + LN2
    float*    tok2 = (float*)(ws + o_tok2);
    _Float16* y2   = (_Float16*)(ws + o_y2);
    lnscale_kernel<<<M2 / 8, 256, 0, stream>>>(acc, norm2_w, norm2_b, tok2, y2);

    // 6. fc1 + bias + GELU (fused epilogue)
    float* h1 = (float*)(ws + o_h1);
    gemm_kernel<MLP, CDIM, 0, 1><<<dim3(3, M2 / 128), 256, 0, stream>>>(
        y2, f116, h1, nullptr, nullptr, nullptr, fc1_b, nullptr);

    // 7. depthwise conv + GELU + residual -> f16
    _Float16* h2 = (_Float16*)(ws + o_h2);
    dwconv_kernel<<<(unsigned)(((size_t)M2 * MLP + 255) / 256), 256, 0, stream>>>(
        h1, dw_w, dw_b, h2);

    // 8. fc2 + bias + residual tok2, fused NCHW-transposed store to d_out
    gemm_kernel<CDIM, MLP, 2, 0><<<dim3(2, M2 / 128), 256, 0, stream>>>(
        h2, f216, (float*)d_out, nullptr, nullptr, nullptr, fc2_b, tok2);
}
